// CLEAR_45561013076524
// MI455X (gfx1250) — compile-verified, matrix operand order checked by reading, not run on hardware
//
#include <hip/hip_runtime.h>

// ---------------------------------------------------------------------------
// CDNA5 (gfx1250) implementation. Wave32. Matrix ops via
// __builtin_amdgcn_wmma_f32_16x16x32_f16 (16x16x32, f16 in / f32 acc).
// Wave tile: 32x64 (2 row-frags x 4 col-frags), 2-deep software pipeline.
// __launch_bounds__(256, 1): run at low occupancy with a large VGPR budget so
// the 8 f32 accumulators + double-buffered fragments stay in registers
// (round-2 build spilled accumulators to scratch inside the WMMA loop).
// ---------------------------------------------------------------------------

typedef __attribute__((ext_vector_type(16))) _Float16 v16h;
typedef __attribute__((ext_vector_type(8)))  float    v8f;

union FragH { v16h h; unsigned int u[8]; };

// Per ISA 7.12.2 (16-bit A 16x32 / K-contiguous Bt): lanes 0-15 hold K 0..7 and
// 16..23, lanes 16-31 hold K 8..15 and 24..31; dword j -> b4 + (j&3) + (j>>2)*8.
__device__ __forceinline__ void load_frag(const unsigned int* __restrict__ p, int b4, FragH& f) {
#pragma unroll
  for (int j = 0; j < 8; ++j)
    f.u[j] = p[b4 + (j & 3) + ((j >> 2) << 3)];
}

__device__ __forceinline__ void zero_frag(FragH& f) {
#pragma unroll
  for (int j = 0; j < 8; ++j) f.u[j] = 0u;
}

// 8 back-to-back WMMAs: 2 row-frags x 4 col-frags.
__device__ __forceinline__ void mma_step(const FragH fa[2], const FragH fb[4], v8f acc[2][4]) {
#pragma unroll
  for (int t = 0; t < 4; ++t) {
#pragma unroll
    for (int r = 0; r < 2; ++r)
      acc[r][t] = __builtin_amdgcn_wmma_f32_16x16x32_f16(
          false, fa[r].h, false, fb[t].h, (short)0, acc[r][t], false, false);
  }
}

// ---------------------------------------------------------------------------
// Generic WMMA GEMM:  out[M,N] = A[M,K](f16) * Bt[N,K](f16)^T  (+bias, relu)
// block = 256 threads = 8 waves; wave -> 32x64 tile; block -> 256x64 tile.
// grid = (M/256, N/64). M % 256 == 0, N % 64 == 0, K % 32 == 0.
// ---------------------------------------------------------------------------
__global__ void __launch_bounds__(256, 1)
gemm_wmma_f16(const _Float16* __restrict__ A,
              const _Float16* __restrict__ Bt,
              const float* __restrict__ bias,
              float* __restrict__ outF,
              _Float16* __restrict__ outH,
              int M, int N, int K, int relu) {
  const int lane = threadIdx.x & 31;
  const int wave = threadIdx.x >> 5;
  const int row0 = blockIdx.x * 256 + wave * 32;
  const int n0   = blockIdx.y * 64;
  const int b4   = (lane & 16) ? 4 : 0;

  const unsigned int* pa[2];
  const unsigned int* pb[4];
#pragma unroll
  for (int r = 0; r < 2; ++r)
    pa[r] = (const unsigned int*)(A + (size_t)(row0 + r * 16 + (lane & 15)) * K);
#pragma unroll
  for (int t = 0; t < 4; ++t)
    pb[t] = (const unsigned int*)(Bt + (size_t)(n0 + t * 16 + (lane & 15)) * K);

  v8f zero8 = {0.f,0.f,0.f,0.f,0.f,0.f,0.f,0.f};
  v8f acc[2][4] = {{zero8, zero8, zero8, zero8}, {zero8, zero8, zero8, zero8}};

  const int STEPS = K >> 5;           // 32 halfs (16 dwords) per step

  FragH fa0[2], fb0[4], fa1[2], fb1[4];

  auto load_step_g = [&](int s, FragH fa[2], FragH fb[4]) {
    const int ko = s << 4;
#pragma unroll
    for (int r = 0; r < 2; ++r) load_frag(pa[r] + ko, b4, fa[r]);
#pragma unroll
    for (int t = 0; t < 4; ++t) load_frag(pb[t] + ko, b4, fb[t]);
  };

  load_step_g(0, fa0, fb0);
  for (int s = 0; s < STEPS; s += 2) {
    if (s + 1 < STEPS) load_step_g(s + 1, fa1, fb1);
    __builtin_prefetch(pa[0] + ((s + 2) << 4), 0, 1);  // global_prefetch_b8
    mma_step(fa0, fb0, acc);
    if (s + 2 < STEPS) load_step_g(s + 2, fa0, fb0);
    if (s + 1 < STEPS) mma_step(fa1, fb1, acc);
  }

#pragma unroll
  for (int r = 0; r < 2; ++r) {
    const int rbase = row0 + r * 16 + ((lane & 16) ? 8 : 0);
#pragma unroll
    for (int t = 0; t < 4; ++t) {
      const int col = n0 + t * 16 + (lane & 15);
      const float bv = bias ? bias[col] : 0.0f;
#pragma unroll
      for (int i = 0; i < 8; ++i) {
        float v = acc[r][t][i] + bv;
        if (relu) v = fmaxf(v, 0.0f);
        const size_t off = (size_t)(rbase + i) * N + col;
        if (outF) outF[off] = v;
        if (outH) outH[off] = (_Float16)v;
      }
    }
  }
}

// ---------------------------------------------------------------------------
// 3x3 conv as implicit GEMM (9-shift accumulation), output spatial 8x8.
// X: (B*Hin*Win, Cin) f16.  Wr: (9, Cout, Cin) f16.  Y: (B*64, Cout) f16, ReLU.
// grid = (B*64/256, Cout/64). Cin = 32*CSTEPS, Cout % 64 == 0.
// Flattened (tap, K-chunk) step sequence, 2-deep pipeline, 32x64 wave tile.
// ---------------------------------------------------------------------------
template <int CSTEPS>
__global__ void __launch_bounds__(256, 1)
conv3x3_wmma(const _Float16* __restrict__ X,
             const _Float16* __restrict__ Wr,
             const float* __restrict__ bias,
             _Float16* __restrict__ Y,
             int Cout, int Hin, int Win, int stride) {
  constexpr int Cin   = CSTEPS * 32;
  constexpr int STEPS = 9 * CSTEPS;

  const int lane = threadIdx.x & 31;
  const int wave = threadIdx.x >> 5;
  const int row0 = blockIdx.x * 256 + wave * 32;
  const int n0   = blockIdx.y * 64;
  const int b4   = (lane & 16) ? 4 : 0;
  const int HinWin = Hin * Win;

  int bimg[2], oy[2], ox[2];
#pragma unroll
  for (int r = 0; r < 2; ++r) {
    const int m = row0 + r * 16 + (lane & 15);
    bimg[r] = m >> 6;          // Hout*Wout == 64
    const int hw = m & 63;
    oy[r] = hw >> 3;           // Wout == 8
    ox[r] = hw & 7;
  }

  auto load_a = [&](int s, FragH fa[2]) {
    const int j  = s / CSTEPS;
    const int kc = (s - j * CSTEPS) * 32;
    const int dy = j / 3, dx = j - dy * 3;
#pragma unroll
    for (int r = 0; r < 2; ++r) {
      const int iy = oy[r] * stride + dy - 1;
      const int ix = ox[r] * stride + dx - 1;
      const bool valid = (iy >= 0) & (iy < Hin) & (ix >= 0) & (ix < Win);
      if (valid) {
        const unsigned int* p = (const unsigned int*)(
            X + (size_t)(bimg[r] * HinWin + iy * Win + ix) * Cin + kc);
        load_frag(p, b4, fa[r]);
      } else {
        zero_frag(fa[r]);
      }
    }
  };
  auto load_b = [&](int s, FragH fb[4]) {
    const int j  = s / CSTEPS;
    const int kc = (s - j * CSTEPS) * 32;
#pragma unroll
    for (int t = 0; t < 4; ++t) {
      const unsigned int* p = (const unsigned int*)(
          Wr + ((size_t)j * Cout + n0 + t * 16 + (lane & 15)) * Cin + kc);
      load_frag(p, b4, fb[t]);
    }
  };

  v8f zero8 = {0.f,0.f,0.f,0.f,0.f,0.f,0.f,0.f};
  v8f acc[2][4] = {{zero8, zero8, zero8, zero8}, {zero8, zero8, zero8, zero8}};

  FragH fa0[2], fb0[4], fa1[2], fb1[4];
  load_a(0, fa0); load_b(0, fb0);
  for (int s = 0; s < STEPS; s += 2) {
    if (s + 1 < STEPS) { load_a(s + 1, fa1); load_b(s + 1, fb1); }
    mma_step(fa0, fb0, acc);
    if (s + 2 < STEPS) { load_a(s + 2, fa0); load_b(s + 2, fb0); }
    if (s + 1 < STEPS) mma_step(fa1, fb1, acc);
  }

#pragma unroll
  for (int r = 0; r < 2; ++r) {
    const int rbase = row0 + r * 16 + ((lane & 16) ? 8 : 0);
#pragma unroll
    for (int t = 0; t < 4; ++t) {
      const int col = n0 + t * 16 + (lane & 15);
      const float bv = bias[col];
#pragma unroll
      for (int i = 0; i < 8; ++i) {
        float v = fmaxf(acc[r][t][i] + bv, 0.0f);
        Y[(size_t)(rbase + i) * Cout + col] = (_Float16)v;
      }
    }
  }
}

// ---------------------------------------------------------------------------
// conv1: 3ch 5x5 stride2 pad1 on 32x32 -> (B*225, 32) f16 with ReLU (direct fp32).
// ---------------------------------------------------------------------------
__global__ void conv1_direct(const float* __restrict__ x, const float* __restrict__ w,
                             const float* __restrict__ bias, _Float16* __restrict__ h1,
                             int total) {
  int idx = blockIdx.x * blockDim.x + threadIdx.x;
  if (idx >= total) return;
  const int c   = idx & 31;
  const int pos = idx >> 5;
  const int ox  = pos % 15;
  const int t2  = pos / 15;
  const int oy  = t2 % 15;
  const int b   = t2 / 15;
  const int iy0 = oy * 2 - 1, ix0 = ox * 2 - 1;
  float s = bias[c];
#pragma unroll
  for (int ci = 0; ci < 3; ++ci) {
#pragma unroll
    for (int ky = 0; ky < 5; ++ky) {
      const int iy = iy0 + ky;
      if (iy < 0 || iy >= 32) continue;
#pragma unroll
      for (int kx = 0; kx < 5; ++kx) {
        const int ix = ix0 + kx;
        if (ix < 0 || ix >= 32) continue;
        s += x[((size_t)(b * 3 + ci) * 32 + iy) * 32 + ix] *
             w[((c * 3 + ci) * 5 + ky) * 5 + kx];
      }
    }
  }
  h1[(size_t)pos * 32 + c] = (_Float16)fmaxf(s, 0.0f);
}

// OIHW (Cout,Cin,3,3) f32 -> (9, Cout, Cin) f16
__global__ void repack_w3x3(const float* __restrict__ src, _Float16* __restrict__ dst,
                            int Cout, int Cin) {
  int idx = blockIdx.x * blockDim.x + threadIdx.x;
  int total = Cout * Cin * 9;
  if (idx >= total) return;
  const int j  = idx % 9;
  const int r  = idx / 9;
  const int ci = r % Cin;
  const int co = r / Cin;
  dst[((size_t)j * Cout + co) * Cin + ci] = (_Float16)src[idx];
}

// h4 (B*64, 256) f16 -> encA (B, 16384) f16 in NCHW-flatten order k = c*64 + hw
__global__ void repack_h4(const _Float16* __restrict__ h4, _Float16* __restrict__ encA,
                          int total) {
  int idx = blockIdx.x * blockDim.x + threadIdx.x;
  if (idx >= total) return;
  const int b  = idx >> 14;
  const int k  = idx & 16383;
  const int c  = k >> 6;
  const int hw = k & 63;
  encA[idx] = h4[(((size_t)b << 6) + hw) * 256 + c];
}

// dst[c*R + r] = (f16) src[r*C + c]   (f32 (R,C) -> f16 (C,R))
__global__ void transpose_f32_to_f16(const float* __restrict__ src, _Float16* __restrict__ dst,
                                     int R, int C) {
  int idx = blockIdx.x * blockDim.x + threadIdx.x;
  if (idx >= R * C) return;
  const int r = idx / C;
  const int c = idx - r * C;
  dst[(size_t)c * R + r] = (_Float16)src[idx];
}

__global__ void f32_to_f16(const float* __restrict__ s, _Float16* __restrict__ d, int n) {
  int idx = blockIdx.x * blockDim.x + threadIdx.x;
  if (idx < n) d[idx] = (_Float16)s[idx];
}

// row-wise squared L2 norm, D == blockDim.x == 256
__global__ void rownorm256(const float* __restrict__ Z, float* __restrict__ out) {
  __shared__ float sb[256];
  const int r = blockIdx.x, t = threadIdx.x;
  float v = Z[(size_t)r * 256 + t];
  sb[t] = v * v;
  __syncthreads();
  for (int o = 128; o > 0; o >>= 1) {
    if (t < o) sb[t] += sb[t + o];
    __syncthreads();
  }
  if (t == 0) out[r] = sb[0];
}

// d_total -> softmax(-d/temp) * sigmoid(gate) -> renormalize -> f16 weights
__global__ void dist_softmax_gate(const float* __restrict__ zn, const float* __restrict__ pn,
                                  const float* __restrict__ gn, const float* __restrict__ dotP,
                                  const float* __restrict__ dotG, const float* __restrict__ temp_raw,
                                  const float* __restrict__ gate, _Float16* __restrict__ wh) {
  __shared__ float sb[256];
  const int b = blockIdx.x, j = threadIdx.x;
  const float znb = zn[b];
  const float dP = znb + pn[j] - 2.0f * dotP[(size_t)b * 256 + j];
  const float dG = znb + gn[j] - 2.0f * dotG[(size_t)b * 256 + j];
  const float d  = sqrtf(fmaxf(dP, 0.0f)) + sqrtf(fmaxf(dG, 0.0f));
  const float temp = 1.0f / (1.0f + expf(-temp_raw[0])) * 0.999f + 0.001f;
  const float s = -d / temp;

  sb[j] = s; __syncthreads();
  for (int o = 128; o > 0; o >>= 1) { if (j < o) sb[j] = fmaxf(sb[j], sb[j + o]); __syncthreads(); }
  const float mx = sb[0]; __syncthreads();

  const float e = expf(s - mx);
  sb[j] = e; __syncthreads();
  for (int o = 128; o > 0; o >>= 1) { if (j < o) sb[j] += sb[j + o]; __syncthreads(); }
  const float sumE = sb[0]; __syncthreads();

  const float t = (e / sumE) * (1.0f / (1.0f + expf(-gate[j])));
  sb[j] = t; __syncthreads();
  for (int o = 128; o > 0; o >>= 1) { if (j < o) sb[j] += sb[j + o]; __syncthreads(); }
  const float sumT = sb[0];

  wh[(size_t)b * 256 + j] = (_Float16)(t / (sumT + 1e-8f));
}

// Fold the 4 head-blocks of clf_w: clfsum[d][c] = sum_h clf_w[h*256+d][c]
// (attention over a single node is exactly identity -> fused == tile(blended,4))
__global__ void clfsum_kernel(const float* __restrict__ clf_w, float* __restrict__ clfsum) {
  int idx = blockIdx.x * blockDim.x + threadIdx.x;
  if (idx >= 2560) return;
  const int d = idx / 10, c = idx % 10;
  float s = 0.0f;
#pragma unroll
  for (int h = 0; h < 4; ++h) s += clf_w[((h << 8) + d) * 10 + c];
  clfsum[idx] = s;
}

// logits[b][c] = sum_d blended[b][d] * clfsum[d][c] + clf_b[c]
__global__ void logits_kernel(const float* __restrict__ blended, const float* __restrict__ clfsum,
                              const float* __restrict__ clf_b, float* __restrict__ out) {
  __shared__ float accs[10];
  const int b = blockIdx.x, t = threadIdx.x;
  if (t < 10) accs[t] = 0.0f;
  __syncthreads();
  const float zb = blended[(size_t)b * 256 + t];
#pragma unroll
  for (int c = 0; c < 10; ++c) atomicAdd(&accs[c], zb * clfsum[t * 10 + c]);
  __syncthreads();
  if (t < 10) out[b * 10 + t] = accs[t] + clf_b[t];
}

// ---------------------------------------------------------------------------
extern "C" void kernel_launch(void* const* d_in, const int* in_sizes, int n_in,
                              void* d_out, int out_size, void* d_ws, size_t ws_size,
                              hipStream_t stream) {
  (void)in_sizes; (void)n_in; (void)out_size; (void)ws_size;
  const float* x        = (const float*)d_in[0];
  const float* conv1_w  = (const float*)d_in[2];
  const float* conv1_b  = (const float*)d_in[3];
  const float* conv2_w  = (const float*)d_in[4];
  const float* conv2_b  = (const float*)d_in[5];
  const float* conv3_w  = (const float*)d_in[6];
  const float* conv3_b  = (const float*)d_in[7];
  const float* conv4_w  = (const float*)d_in[8];
  const float* conv4_b  = (const float*)d_in[9];
  const float* enc_w    = (const float*)d_in[10];
  const float* enc_b    = (const float*)d_in[11];
  const float* clf_w    = (const float*)d_in[18];
  const float* clf_b    = (const float*)d_in[19];
  const float* nfc_w    = (const float*)d_in[20];
  const float* nfc_b    = (const float*)d_in[21];
  const float* protos   = (const float*)d_in[22];
  const float* gridp    = (const float*)d_in[23];
  const float* temp_raw = (const float*)d_in[24];
  const float* gate     = (const float*)d_in[25];
  float* out = (float*)d_out;

  const int Bn = 4096;

  // bump allocator over d_ws
  char* wsp = (char*)d_ws;
  size_t off = 0;
  auto alloc = [&](size_t bytes) -> void* {
    void* p = wsp + off;
    off += (bytes + 255) & ~(size_t)255;
    return p;
  };

  _Float16* h1h   = (_Float16*)alloc((size_t)Bn * 225 * 32 * 2);
  _Float16* h2h   = (_Float16*)alloc((size_t)Bn * 64 * 64 * 2);
  _Float16* h3h   = (_Float16*)alloc((size_t)Bn * 64 * 128 * 2);
  _Float16* h4h   = (_Float16*)alloc((size_t)Bn * 64 * 256 * 2);
  _Float16* encA  = (_Float16*)alloc((size_t)Bn * 16384 * 2);
  _Float16* encBt = (_Float16*)alloc((size_t)256 * 16384 * 2);
  _Float16* Wr2   = (_Float16*)alloc((size_t)9 * 64 * 32 * 2);
  _Float16* Wr3   = (_Float16*)alloc((size_t)9 * 128 * 64 * 2);
  _Float16* Wr4   = (_Float16*)alloc((size_t)9 * 256 * 128 * 2);
  _Float16* nfcBt = (_Float16*)alloc((size_t)256 * 256 * 2);
  _Float16* protoH= (_Float16*)alloc((size_t)256 * 256 * 2);
  _Float16* gridH = (_Float16*)alloc((size_t)256 * 256 * 2);
  _Float16* protoT= (_Float16*)alloc((size_t)256 * 256 * 2);
  _Float16* z0h   = (_Float16*)alloc((size_t)Bn * 256 * 2);
  float*    zf    = (float*)   alloc((size_t)Bn * 256 * 4);
  _Float16* zh    = (_Float16*)alloc((size_t)Bn * 256 * 2);
  float*    dotP  = (float*)   alloc((size_t)Bn * 256 * 4);
  float*    dotG  = (float*)   alloc((size_t)Bn * 256 * 4);
  float*    zn    = (float*)   alloc((size_t)Bn * 4);
  float*    pn    = (float*)   alloc((size_t)256 * 4);
  float*    gn    = (float*)   alloc((size_t)256 * 4);
  _Float16* wh    = (_Float16*)alloc((size_t)Bn * 256 * 2);
  float*    blendf= (float*)   alloc((size_t)Bn * 256 * 4);
  float*    clfsum= (float*)   alloc((size_t)2560 * 4);

  const dim3 blk(256);

  // ---- weight / constant repacks ----
  repack_w3x3<<<dim3((64 * 32 * 9 + 255) / 256), blk, 0, stream>>>(conv2_w, Wr2, 64, 32);
  repack_w3x3<<<dim3((128 * 64 * 9 + 255) / 256), blk, 0, stream>>>(conv3_w, Wr3, 128, 64);
  repack_w3x3<<<dim3((256 * 128 * 9 + 255) / 256), blk, 0, stream>>>(conv4_w, Wr4, 256, 128);
  transpose_f32_to_f16<<<dim3((16384 * 256) / 256), blk, 0, stream>>>(enc_w, encBt, 16384, 256);
  transpose_f32_to_f16<<<dim3((256 * 256) / 256), blk, 0, stream>>>(nfc_w, nfcBt, 256, 256);
  transpose_f32_to_f16<<<dim3((256 * 256) / 256), blk, 0, stream>>>(protos, protoT, 256, 256);
  f32_to_f16<<<dim3((256 * 256) / 256), blk, 0, stream>>>(protos, protoH, 256 * 256);
  f32_to_f16<<<dim3((256 * 256) / 256), blk, 0, stream>>>(gridp, gridH, 256 * 256);
  rownorm256<<<dim3(256), blk, 0, stream>>>(protos, pn);
  rownorm256<<<dim3(256), blk, 0, stream>>>(gridp, gn);
  clfsum_kernel<<<dim3(10), blk, 0, stream>>>(clf_w, clfsum);

  // ---- conv tower ----
  conv1_direct<<<dim3((Bn * 225 * 32) / 256), blk, 0, stream>>>(x, conv1_w, conv1_b, h1h,
                                                                Bn * 225 * 32);
  conv3x3_wmma<1><<<dim3(Bn * 64 / 256, 1), blk, 0, stream>>>(h1h, Wr2, conv2_b, h2h,
                                                              64, 15, 15, 2);
  conv3x3_wmma<2><<<dim3(Bn * 64 / 256, 2), blk, 0, stream>>>(h2h, Wr3, conv3_b, h3h,
                                                              128, 8, 8, 1);
  conv3x3_wmma<4><<<dim3(Bn * 64 / 256, 4), blk, 0, stream>>>(h3h, Wr4, conv4_b, h4h,
                                                              256, 8, 8, 1);

  // ---- encoder + node_fc ----
  repack_h4<<<dim3((Bn * 16384) / 256), blk, 0, stream>>>(h4h, encA, Bn * 16384);
  gemm_wmma_f16<<<dim3(Bn / 256, 4), blk, 0, stream>>>(encA, encBt, enc_b,
                                                       (float*)nullptr, z0h,
                                                       Bn, 256, 16384, 0);
  gemm_wmma_f16<<<dim3(Bn / 256, 4), blk, 0, stream>>>(z0h, nfcBt, nfc_b,
                                                       zf, zh, Bn, 256, 256, 0);

  // ---- cdist + gated softmax + blend ----
  gemm_wmma_f16<<<dim3(Bn / 256, 4), blk, 0, stream>>>(zh, protoH, (const float*)nullptr,
                                                       dotP, (_Float16*)nullptr,
                                                       Bn, 256, 256, 0);
  gemm_wmma_f16<<<dim3(Bn / 256, 4), blk, 0, stream>>>(zh, gridH, (const float*)nullptr,
                                                       dotG, (_Float16*)nullptr,
                                                       Bn, 256, 256, 0);
  rownorm256<<<dim3(Bn), blk, 0, stream>>>(zf, zn);
  dist_softmax_gate<<<dim3(Bn), blk, 0, stream>>>(zn, pn, gn, dotP, dotG, temp_raw, gate, wh);
  gemm_wmma_f16<<<dim3(Bn / 256, 4), blk, 0, stream>>>(wh, protoT, (const float*)nullptr,
                                                       blendf, (_Float16*)nullptr,
                                                       Bn, 256, 256, 0);

  // ---- classifier (attention over 1 node == identity; heads folded) ----
  logits_kernel<<<dim3(Bn), blk, 0, stream>>>(blendf, clfsum, clf_b, out);
}
